// GCN_88931592831165
// MI455X (gfx1250) — matrix operand order repaired and checked by reading, not compiled, not run on hardware
//
#include <hip/hip_runtime.h>

typedef __attribute__((ext_vector_type(16))) _Float16 v16h;
typedef __attribute__((ext_vector_type(8)))  float    v8f;

#define K_DIM 128

// ---------------------------------------------------------------------------
// f32 -> f16 weight conversion (tiny: 128x128 / 128x64)
// ---------------------------------------------------------------------------
__global__ void cvt_f32_to_f16(const float* __restrict__ src,
                               _Float16* __restrict__ dst, int n) {
    int i = blockIdx.x * blockDim.x + threadIdx.x;
    if (i < n) dst[i] = (_Float16)src[i];
}

// ---------------------------------------------------------------------------
// WMMA GEMM: C[M,N] = act(A[M,128]) @ Bh[128,N] (+ bias)
// One wave computes a 16 x N stripe, reusing the A fragment across N tiles.
// NT = N/16.  RELU_IN applies relu to A during load (fuses relu(agg) @ W).
// ---------------------------------------------------------------------------
template <int NT, bool RELU_IN, bool BIAS>
__global__ __launch_bounds__(256) void gcn_gemm_wmma(
    const float* __restrict__ A, const _Float16* __restrict__ Bh,
    const float* __restrict__ bias, float* __restrict__ C, int M) {
    const int N = NT * 16;
    const int waveId = (blockIdx.x * blockDim.x + threadIdx.x) >> 5;
    const int mtiles = M >> 4;
    if (waveId >= mtiles) return;              // wave-uniform: EXEC stays all-1s

    const int lane = threadIdx.x & 31;
    const int r    = lane & 15;                // row within tile / N col at store
    const int h    = lane >> 4;                // half-wave selector
    const int m0   = waveId * 16;

    v8f acc[NT];
#pragma unroll
    for (int t = 0; t < NT; ++t) acc[t] = (v8f){0.f, 0.f, 0.f, 0.f, 0.f, 0.f, 0.f, 0.f};

    const float* arow = A + (size_t)(m0 + r) * K_DIM;

    for (int k0 = 0; k0 < K_DIM; k0 += 32) {
        // ---- A fragment: 16x32 f16. Lane holds K = k0+8h+[0,8) and k0+16+8h+[0,8)
        const float4* p0 = (const float4*)(arow + k0 + h * 8);
        const float4* p1 = (const float4*)(arow + k0 + 16 + h * 8);
        float4 f0 = p0[0], f1 = p0[1], f2 = p1[0], f3 = p1[1];
        if (RELU_IN) {
            f0.x = fmaxf(f0.x, 0.f); f0.y = fmaxf(f0.y, 0.f); f0.z = fmaxf(f0.z, 0.f); f0.w = fmaxf(f0.w, 0.f);
            f1.x = fmaxf(f1.x, 0.f); f1.y = fmaxf(f1.y, 0.f); f1.z = fmaxf(f1.z, 0.f); f1.w = fmaxf(f1.w, 0.f);
            f2.x = fmaxf(f2.x, 0.f); f2.y = fmaxf(f2.y, 0.f); f2.z = fmaxf(f2.z, 0.f); f2.w = fmaxf(f2.w, 0.f);
            f3.x = fmaxf(f3.x, 0.f); f3.y = fmaxf(f3.y, 0.f); f3.z = fmaxf(f3.z, 0.f); f3.w = fmaxf(f3.w, 0.f);
        }
        v16h a;
        a[0]  = (_Float16)f0.x; a[1]  = (_Float16)f0.y; a[2]  = (_Float16)f0.z; a[3]  = (_Float16)f0.w;
        a[4]  = (_Float16)f1.x; a[5]  = (_Float16)f1.y; a[6]  = (_Float16)f1.z; a[7]  = (_Float16)f1.w;
        a[8]  = (_Float16)f2.x; a[9]  = (_Float16)f2.y; a[10] = (_Float16)f2.z; a[11] = (_Float16)f2.w;
        a[12] = (_Float16)f3.x; a[13] = (_Float16)f3.y; a[14] = (_Float16)f3.z; a[15] = (_Float16)f3.w;

        // ---- B fragments: 32x16 f16 tile; lane <-> N col r, K = k0+8h+[0,8) and k0+16+8h+[0,8)
        const int kk = k0 + 8 * h;
        const _Float16* brow = Bh + (size_t)kk * N + r;
#pragma unroll
        for (int t = 0; t < NT; ++t) {
            v16h b;
#pragma unroll
            for (int s = 0; s < 8; ++s) {
                b[s]     = brow[(size_t)s * N + t * 16];
                b[8 + s] = brow[(size_t)(16 + s) * N + t * 16];
            }
            acc[t] = __builtin_amdgcn_wmma_f32_16x16x32_f16(
                false, a, false, b, (short)0, acc[t], false, false);
        }
    }

    // ---- store: VGPR j -> M = m0 + j + 8h, N = t*16 + r
#pragma unroll
    for (int t = 0; t < NT; ++t) {
        const int n  = t * 16 + r;
        const float bv = BIAS ? bias[n] : 0.f;
#pragma unroll
        for (int j = 0; j < 8; ++j) {
            const int m = m0 + j + 8 * h;
            C[(size_t)m * N + n] = acc[t][j] + bv;
        }
    }
}

// ---------------------------------------------------------------------------
// Edge scatter-add: agg[rows[e], :] += support[cols[e], :], F = 128.
// One wave per edge; lane handles 4 contiguous floats (float4 gather +
// 4x global_atomic_add_f32). agg (25.6 MB) is L2-resident on MI455X.
// ---------------------------------------------------------------------------
__global__ __launch_bounds__(256) void edge_scatter_add(
    const int* __restrict__ rows, const int* __restrict__ cols,
    const float* __restrict__ support, float* __restrict__ agg, int nEdges) {
    const int e = (blockIdx.x * blockDim.x + threadIdx.x) >> 5;
    if (e >= nEdges) return;
    const int lane = threadIdx.x & 31;
    const int dstRow = rows[e];
    const int srcRow = cols[e];
    const float4 v = ((const float4*)(support + (size_t)srcRow * K_DIM))[lane];
    float* d = agg + (size_t)dstRow * K_DIM + lane * 4;
    atomicAdd(d + 0, v.x);
    atomicAdd(d + 1, v.y);
    atomicAdd(d + 2, v.z);
    atomicAdd(d + 3, v.w);
}

// ---------------------------------------------------------------------------
extern "C" void kernel_launch(void* const* d_in, const int* in_sizes, int n_in,
                              void* d_out, int out_size, void* d_ws, size_t ws_size,
                              hipStream_t stream) {
    const int*   edge_index = (const int*)d_in[0];   // [2, E] int32
    const float* features   = (const float*)d_in[1]; // [N, 128]
    const float* W0         = (const float*)d_in[2]; // [128, 128]
    const float* W1         = (const float*)d_in[3]; // [128, 128]
    const float* W_out      = (const float*)d_in[4]; // [128, 64]
    const float* b_out      = (const float*)d_in[5]; // [64]
    float*       out        = (float*)d_out;         // [N, 64]

    const int E = in_sizes[0] / 2;
    const int N = in_sizes[1] / K_DIM;
    const int OUTF = in_sizes[5];                    // 64

    // workspace carve-out (256B aligned)
    char* ws = (char*)d_ws;
    size_t off = 0;
    auto carve = [&](size_t bytes) {
        off = (off + 255) & ~(size_t)255;
        void* p = ws + off;
        off += bytes;
        return p;
    };
    float*     support = (float*)carve((size_t)N * K_DIM * sizeof(float));
    float*     agg     = (float*)carve((size_t)N * K_DIM * sizeof(float));
    _Float16*  W0h     = (_Float16*)carve((size_t)K_DIM * K_DIM * sizeof(_Float16));
    _Float16*  W1h     = (_Float16*)carve((size_t)K_DIM * K_DIM * sizeof(_Float16));
    _Float16*  Wouth   = (_Float16*)carve((size_t)K_DIM * OUTF * sizeof(_Float16));

    // weight conversion (tiny)
    {
        int n = K_DIM * K_DIM;
        cvt_f32_to_f16<<<(n + 255) / 256, 256, 0, stream>>>(W0, W0h, n);
        cvt_f32_to_f16<<<(n + 255) / 256, 256, 0, stream>>>(W1, W1h, n);
        int m = K_DIM * OUTF;
        cvt_f32_to_f16<<<(m + 255) / 256, 256, 0, stream>>>(W_out, Wouth, m);
    }

    const int mtiles     = N / 16;                    // 3125
    const int gemmBlocks = (mtiles + 7) / 8;          // 8 waves / block
    const int scatBlocks = (E + 7) / 8;               // 8 edges / block
    const size_t aggBytes = (size_t)N * K_DIM * sizeof(float);

    const int* rows = edge_index;       // edge_index[0]: segment ids (dst)
    const int* cols = edge_index + E;   // edge_index[1]: gather ids (src)

    // Layer 1: support = X @ W0 ; agg = A @ support ; relu fused into next load
    gcn_gemm_wmma<8, false, false><<<gemmBlocks, 256, 0, stream>>>(
        features, W0h, nullptr, support, N);
    hipMemsetAsync(agg, 0, aggBytes, stream);
    edge_scatter_add<<<scatBlocks, 256, 0, stream>>>(rows, cols, support, agg, E);

    // Layer 2: support = relu(agg) @ W1 ; agg = A @ support
    gcn_gemm_wmma<8, true, false><<<gemmBlocks, 256, 0, stream>>>(
        agg, W1h, nullptr, support, N);
    // reuse support as new agg? No: scatter reads support. Zero a fresh agg pass:
    hipMemsetAsync(agg, 0, 0, stream); // no-op placeholder removed below
    // NOTE: agg was consumed by the GEMM above; safe to reuse it for layer-2 agg.
    hipMemsetAsync(agg, 0, aggBytes, stream);
    edge_scatter_add<<<scatBlocks, 256, 0, stream>>>(rows, cols, support, agg, E);

    // Head: out = relu(agg) @ W_out + b_out
    gcn_gemm_wmma<4, true, true><<<gemmBlocks, 256, 0, stream>>>(
        agg, Wouth, b_out, out, N);
}
